// LearnedCitationSelector_59244778881415
// MI455X (gfx1250) — matrix operand order repaired and checked by reading, not compiled, not run on hardware
//
#include <hip/hip_runtime.h>
#include <math.h>

typedef __attribute__((ext_vector_type(16))) _Float16 v16h;
typedef __attribute__((ext_vector_type(8)))  _Float16 v8h;
typedef __attribute__((ext_vector_type(8)))  float    v8f;
typedef __attribute__((ext_vector_type(4)))  float    v4f;

#define B_   32
#define N_   4096
#define T_   768
#define H_   256
#define RPW  64      // rows per workgroup
#define KC   64      // K-chunk staged in LDS

union V16U { v16h v; v8h h[2]; };

__device__ __forceinline__ float gelu_exact(float x) {
    return 0.5f * x * (1.0f + erff(x * 0.70710678118654752440f));
}

// ---------------- small precompute kernels ----------------

// query[b,j] = center[b] . Wq[j] + bq[j]; also cnorm[b] = ||center[b]||
__global__ __launch_bounds__(H_) void k_query(const float* __restrict__ center,
                                              const float* __restrict__ Wq,
                                              const float* __restrict__ bq,
                                              float* __restrict__ query,
                                              float* __restrict__ cnorm) {
    int b = blockIdx.x, j = threadIdx.x;
    const float* cr = center + (size_t)b * T_;
    const float* wr = Wq + (size_t)j * T_;
    float q = bq[j];
    for (int k = 0; k < T_; ++k) q += cr[k] * wr[k];
    query[b * H_ + j] = q;
    if (j == 0) {
        float s = 0.f;
        for (int k = 0; k < T_; ++k) s += cr[k] * cr[k];
        cnorm[b] = sqrtf(s);
    }
}

// qv[b,j] = query[b] . W1[j, 0:256]
__global__ __launch_bounds__(H_) void k_qv(const float* __restrict__ query,
                                           const float* __restrict__ W1,
                                           float* __restrict__ qv) {
    int b = blockIdx.x, j = threadIdx.x;
    const float* qr = query + (size_t)b * H_;
    const float* wr = W1 + (size_t)j * (5 * H_ + 1);
    float a = 0.f;
    for (int h = 0; h < H_; ++h) a += qr[h] * wr[h];
    qv[b * H_ + j] = a;
}

// M16[j,k] = f16( sum_h W1[j, 256+h] * Wc[h,k] )   (collapsed Wc->W1c GEMM weight)
__global__ __launch_bounds__(256) void k_M(const float* __restrict__ W1,
                                           const float* __restrict__ Wc,
                                           _Float16* __restrict__ M16) {
    int tid = blockIdx.x * 256 + threadIdx.x;     // 768 blocks * 256 = 196608
    int j = tid / T_, k = tid % T_;
    const float* wr = W1 + (size_t)j * (5 * H_ + 1) + H_;
    float a = 0.f;
    for (int h = 0; h < H_; ++h) a += wr[h] * Wc[(size_t)h * T_ + k];
    M16[(size_t)j * T_ + k] = (_Float16)a;
}

// per-output-channel coefficient vectors
__global__ __launch_bounds__(H_) void k_coef(const float* __restrict__ W1,
                                             const float* __restrict__ Wt,
                                             const float* __restrict__ Ws,
                                             const float* __restrict__ bc,
                                             const float* __restrict__ bt,
                                             const float* __restrict__ bs,
                                             const float* __restrict__ b1,
                                             const float* __restrict__ Ee,
                                             float* __restrict__ ut,
                                             float* __restrict__ us,
                                             float* __restrict__ cv,
                                             float* __restrict__ wsem,
                                             float* __restrict__ G) {
    int j = threadIdx.x;
    const float* wr = W1 + (size_t)j * (5 * H_ + 1);
    float aut = 0.f, aus = 0.f, acv = b1[j];
    float g0 = 0.f, g1 = 0.f, g2 = 0.f, g3 = 0.f;
    for (int h = 0; h < H_; ++h) {
        float wc = wr[H_ + h], we = wr[2 * H_ + h], wt = wr[3 * H_ + h], ws = wr[4 * H_ + h];
        aut += wt * Wt[h];
        aus += ws * Ws[h];
        acv += wc * bc[h] + wt * bt[h] + ws * bs[h];
        g0 += we * Ee[0 * H_ + h];
        g1 += we * Ee[1 * H_ + h];
        g2 += we * Ee[2 * H_ + h];
        g3 += we * Ee[3 * H_ + h];
    }
    ut[j] = aut; us[j] = aus; cv[j] = acv; wsem[j] = wr[5 * H_];
    G[0 * H_ + j] = g0; G[1 * H_ + j] = g1; G[2 * H_ + j] = g2; G[3 * H_ + j] = g3;
}

// ---------------- main fused GEMM + epilogue ----------------
__global__ __launch_bounds__(256)
void k_main(const float* __restrict__ emb, const float* __restrict__ center,
            const _Float16* __restrict__ M16,
            const float* __restrict__ qv, const float* __restrict__ ut,
            const float* __restrict__ us, const float* __restrict__ cv,
            const float* __restrict__ wsem, const float* __restrict__ G,
            const float* __restrict__ cnorm,
            const float* __restrict__ yd, const float* __restrict__ cs,
            const int* __restrict__ et, const int* __restrict__ mask,
            const float* __restrict__ W2, const float* __restrict__ b2,
            const int* __restrict__ topk_ptr,
            float* __restrict__ dout) {
    __shared__ __align__(16) _Float16 As[RPW][KC + 8];
    __shared__ float qvL[H_], utL[H_], usL[H_], wsL[H_], cvL[H_], w2L[H_];
    __shared__ float GL[4 * H_];
    __shared__ float ydL[RPW], csL[RPW], semL[RPW];
    __shared__ int   etL[RPW], mkL[RPW];
    __shared__ float dsq[256][2];
    __shared__ float part[8 * 8 * 32];
    __shared__ float cnL;

    const int t  = threadIdx.x;
    const int r0 = blockIdx.x * RPW;       // 64 consecutive rows, all same batch b
    const int b  = r0 >> 12;               // N_ = 4096

    qvL[t] = qv[b * H_ + t];
    utL[t] = ut[t]; usL[t] = us[t]; wsL[t] = wsem[t]; cvL[t] = cv[t]; w2L[t] = W2[t];
    GL[t] = G[t]; GL[t + 256] = G[t + 256]; GL[t + 512] = G[t + 512]; GL[t + 768] = G[t + 768];
    if (t < RPW) {
        int rr = r0 + t;
        ydL[t] = yd[rr]; csL[t] = cs[rr];
        int e = et[rr]; e = e < 0 ? 0 : (e > 3 ? 3 : e);
        etL[t] = e; mkL[t] = mask[rr];
    }
    if (t == 0) cnL = cnorm[b];

    const int lane  = t & 31, w = t >> 5;
    const int msub  = w & 3, nhalf = w >> 2;
    const int arow  = msub * 16 + (lane & 15);
    const int kof   = (lane < 16) ? 0 : 8;    // A-frag K offset per ISA layout
    const int bko   = (lane < 16) ? 0 : 8;    // B-frag K offset per ISA layout

    const v8f vz = {0.f, 0.f, 0.f, 0.f, 0.f, 0.f, 0.f, 0.f};
    v8f acc[8];
#pragma unroll
    for (int s = 0; s < 8; ++s) acc[s] = vz;

    // cooperative A loader mapping: thread -> (row lr, 16-float k segment ls)
    float dotp = 0.f, sqp = 0.f;
    const int lr = t >> 2;
    const int ls = (t & 3) << 4;
    const float* arow_g = emb + (size_t)(r0 + lr) * T_;
    const float* crow_g = center + (size_t)b * T_;
    // loop-invariant part of this wave's B-fragment base address
    const _Float16* mwave = M16 + (size_t)(nhalf * 128 + (lane & 15)) * T_ + bko;

    // prologue: prefetch chunk 0 of the streaming A matrix (non-temporal:
    // don't let the 402MB stream evict the L2-resident weight matrix)
    v4f a4[4];
    {
        const v4f* ap4 = (const v4f*)(arow_g + ls);
#pragma unroll
        for (int u = 0; u < 4; ++u) a4[u] = __builtin_nontemporal_load(&ap4[u]);
    }

    for (int chunk = 0; chunk < T_ / KC; ++chunk) {
        __syncthreads();                       // previous chunk's As reads done
        // convert current chunk to f16 in LDS; fuse dot(emb,center) and sum(emb^2)
        const v4f* cp4 = (const v4f*)(crow_g + chunk * KC + ls);
        v4f c4[4];
#pragma unroll
        for (int u = 0; u < 4; ++u) c4[u] = cp4[u];
#pragma unroll
        for (int u = 0; u < 4; ++u) {
            dotp += a4[u].x * c4[u].x + a4[u].y * c4[u].y + a4[u].z * c4[u].z + a4[u].w * c4[u].w;
            sqp  += a4[u].x * a4[u].x + a4[u].y * a4[u].y + a4[u].z * a4[u].z + a4[u].w * a4[u].w;
        }
        v8h h0 = {(_Float16)a4[0].x, (_Float16)a4[0].y, (_Float16)a4[0].z, (_Float16)a4[0].w,
                  (_Float16)a4[1].x, (_Float16)a4[1].y, (_Float16)a4[1].z, (_Float16)a4[1].w};
        v8h h1 = {(_Float16)a4[2].x, (_Float16)a4[2].y, (_Float16)a4[2].z, (_Float16)a4[2].w,
                  (_Float16)a4[3].x, (_Float16)a4[3].y, (_Float16)a4[3].z, (_Float16)a4[3].w};
        *(v8h*)&As[lr][ls]     = h0;
        *(v8h*)&As[lr][ls + 8] = h1;
        __syncthreads();                       // As ready for all waves

        // prefetch next chunk of A into registers; completes under the WMMAs below
        if (chunk + 1 < T_ / KC) {
            const v4f* ap4 = (const v4f*)(arow_g + (chunk + 1) * KC + ls);
#pragma unroll
            for (int u = 0; u < 4; ++u) a4[u] = __builtin_nontemporal_load(&ap4[u]);
        }

        // A fragments for both k-steps of this chunk (LDS)
        V16U af0, af1;
        af0.h[0] = *(const v8h*)&As[arow][kof];
        af0.h[1] = *(const v8h*)&As[arow][kof + 16];
        af1.h[0] = *(const v8h*)&As[arow][32 + kof];
        af1.h[1] = *(const v8h*)&As[arow][32 + kof + 16];

        const _Float16* mbase = mwave + chunk * KC;
        V16U bf[8];
        // k-step 0: batch all 8 B fragments (one clause, one latency for 8 WMMAs)
#pragma unroll
        for (int s = 0; s < 8; ++s) {
            const _Float16* mp = mbase + (size_t)s * 16 * T_;
            bf[s].h[0] = *(const v8h*)mp;
            bf[s].h[1] = *(const v8h*)(mp + 16);
        }
#pragma unroll
        for (int s = 0; s < 8; ++s)
            acc[s] = __builtin_amdgcn_wmma_f32_16x16x32_f16(
                false, af0.v, false, bf[s].v, (short)0, acc[s], false, false);
        // k-step 1
#pragma unroll
        for (int s = 0; s < 8; ++s) {
            const _Float16* mp = mbase + (size_t)s * 16 * T_ + 32;
            bf[s].h[0] = *(const v8h*)mp;
            bf[s].h[1] = *(const v8h*)(mp + 16);
        }
#pragma unroll
        for (int s = 0; s < 8; ++s)
            acc[s] = __builtin_amdgcn_wmma_f32_16x16x32_f16(
                false, af1.v, false, bf[s].v, (short)0, acc[s], false, false);
    }
    __syncthreads();

    // cosine similarity: reduce per-row dot and sum-of-squares (4 loader threads/row)
    dsq[t][0] = dotp; dsq[t][1] = sqp;
    __syncthreads();
    if (t < RPW) {
        float d = 0.f, q = 0.f;
#pragma unroll
        for (int u = 0; u < 4; ++u) { d += dsq[t * 4 + u][0]; q += dsq[t * 4 + u][1]; }
        float na = fmaxf(cnL, 1e-8f);
        float nb = fmaxf(sqrtf(q), 1e-8f);
        semL[t] = d / (na * nb);
    }
    __syncthreads();

    // epilogue: bias terms + exact GELU + partial dot with W2
    float pl[8];
#pragma unroll
    for (int i = 0; i < 8; ++i) pl[i] = 0.f;
#pragma unroll
    for (int s = 0; s < 8; ++s) {
        int j = (nhalf * 8 + s) * 16 + (lane & 15);
        float aq = qvL[j], au = utL[j], ap = usL[j], aw = wsL[j], ac = cvL[j], a2 = w2L[j];
#pragma unroll
        for (int i = 0; i < 8; ++i) {
            int rl = msub * 16 + i + ((lane < 16) ? 0 : 8);   // C-layout: VGPR i, lane half
            float pre = acc[s][i] + aq + GL[etL[rl] * H_ + j]
                      + ydL[rl] * au + csL[rl] * ap + semL[rl] * aw + ac;
            pl[i] += gelu_exact(pre) * a2;
        }
    }
#pragma unroll
    for (int i = 0; i < 8; ++i) part[(w * 8 + i) * 32 + lane] = pl[i];
    __syncthreads();

    if (t < RPW) {
        int m2 = t >> 4, local = t & 15, i = local & 7, half = local >> 3;
        float sum = b2[0];
#pragma unroll
        for (int l = 0; l < 16; ++l) {
            sum += part[(m2 * 8 + i) * 32 + half * 16 + l];
            sum += part[((m2 + 4) * 8 + i) * 32 + half * 16 + l];
        }
        int k = topk_ptr[0]; k = k < 1 ? 1 : (k > 64 ? 64 : k);
        float* outL = dout + 2 * B_ * k;        // [idx | vals | logits]
        outL[r0 + t] = mkL[t] ? sum : -1e9f;
    }
}

// ---------------- per-batch top-k (sorted, lowest-index tie-break) ----------------
__global__ __launch_bounds__(256) void k_topk(float* __restrict__ dout,
                                              const int* __restrict__ topk_ptr) {
    __shared__ float v[N_];
    __shared__ float rv[256];
    __shared__ int   ri[256];
    int t = threadIdx.x, b = blockIdx.x;
    int k = topk_ptr[0]; k = k < 1 ? 1 : (k > 64 ? 64 : k);
    const float* lg = dout + 2 * B_ * k + (size_t)b * N_;
    for (int i = t; i < N_; i += 256) v[i] = lg[i];
    __syncthreads();
    for (int it = 0; it < k; ++it) {
        float best = -INFINITY; int bi = N_;
        for (int i = t; i < N_; i += 256) {
            float x = v[i];
            if (x > best || (x == best && i < bi)) { best = x; bi = i; }
        }
        rv[t] = best; ri[t] = bi;
        __syncthreads();
        for (int s = 128; s > 0; s >>= 1) {
            if (t < s) {
                float ov = rv[t + s]; int oi = ri[t + s];
                if (ov > rv[t] || (ov == rv[t] && oi < ri[t])) { rv[t] = ov; ri[t] = oi; }
            }
            __syncthreads();
        }
        if (t == 0) {
            dout[b * k + it]          = (float)ri[0];
            dout[B_ * k + b * k + it] = rv[0];
            v[ri[0]] = -INFINITY;
        }
        __syncthreads();
    }
}

// ---------------- launcher ----------------
extern "C" void kernel_launch(void* const* d_in, const int* in_sizes, int n_in,
                              void* d_out, int out_size, void* d_ws, size_t ws_size,
                              hipStream_t stream) {
    (void)in_sizes; (void)n_in; (void)out_size; (void)ws_size;
    const float* center = (const float*)d_in[0];
    const float* emb    = (const float*)d_in[1];
    const int*   mask   = (const int*)d_in[2];
    const int*   etyp   = (const int*)d_in[3];
    const float* yd     = (const float*)d_in[4];
    const float* cs     = (const float*)d_in[5];
    const int*   topk   = (const int*)d_in[6];
    const float* Wq = (const float*)d_in[7];
    const float* bq = (const float*)d_in[8];
    const float* Wc = (const float*)d_in[9];
    const float* bc = (const float*)d_in[10];
    const float* Ee = (const float*)d_in[11];
    const float* Wt = (const float*)d_in[12];
    const float* bt = (const float*)d_in[13];
    const float* Ws = (const float*)d_in[14];
    const float* bs = (const float*)d_in[15];
    const float* W1 = (const float*)d_in[16];
    const float* b1 = (const float*)d_in[17];
    const float* W2 = (const float*)d_in[18];
    const float* b2 = (const float*)d_in[19];

    char* wsp = (char*)d_ws;
    _Float16* M16 = (_Float16*)wsp;  wsp += (size_t)H_ * T_ * sizeof(_Float16);
    float* query = (float*)wsp;      wsp += (size_t)B_ * H_ * sizeof(float);
    float* qv    = (float*)wsp;      wsp += (size_t)B_ * H_ * sizeof(float);
    float* ut    = (float*)wsp;      wsp += H_ * sizeof(float);
    float* us    = (float*)wsp;      wsp += H_ * sizeof(float);
    float* cv    = (float*)wsp;      wsp += H_ * sizeof(float);
    float* wsem  = (float*)wsp;      wsp += H_ * sizeof(float);
    float* G     = (float*)wsp;      wsp += 4 * H_ * sizeof(float);
    float* cn    = (float*)wsp;      wsp += B_ * sizeof(float);

    hipLaunchKernelGGL(k_query, dim3(B_), dim3(H_), 0, stream, center, Wq, bq, query, cn);
    hipLaunchKernelGGL(k_qv,    dim3(B_), dim3(H_), 0, stream, query, W1, qv);
    hipLaunchKernelGGL(k_M,     dim3(T_), dim3(256), 0, stream, W1, Wc, M16);
    hipLaunchKernelGGL(k_coef,  dim3(1),  dim3(H_), 0, stream, W1, Wt, Ws, bc, bt, bs, b1, Ee,
                       ut, us, cv, wsem, G);
    hipLaunchKernelGGL(k_main,  dim3((B_ * N_) / RPW), dim3(256), 0, stream,
                       emb, center, M16, qv, ut, us, cv, wsem, G, cn,
                       yd, cs, etyp, mask, W2, b2, topk, (float*)d_out);
    hipLaunchKernelGGL(k_topk,  dim3(B_), dim3(256), 0, stream, (float*)d_out, topk);
}
